// LSM_IniReconNet_4681514353454
// MI455X (gfx1250) — compile-verified, matrix operand order checked by reading, not run on hardware
//
#include <hip/hip_runtime.h>

typedef __attribute__((ext_vector_type(16))) _Float16 v16h;
typedef __attribute__((ext_vector_type(8)))  _Float16 v8h;
typedef __attribute__((ext_vector_type(8)))  float    v8f;

#define BS    32
#define SP    256
#define KDIM  1024          // BS*BS
#define NROWS 16384         // B * (H/BS) * (W/BS) = 16*32*32

// GEMM tiling
#define BM 128
#define BN 128
#define BK 32

// ---------------------------------------------------------------------------
// Kernel 1: M[k_out, k_in] = sum_s w_init[k_out, s] * w_sample[s, k_in]
// (fuses the two convolutions into one 1024x1024 matrix, stored fp16)
// ---------------------------------------------------------------------------
__global__ __launch_bounds__(256)
void build_weight(const float* __restrict__ w_sample,
                  const float* __restrict__ w_init,
                  _Float16* __restrict__ Mw) {
    const int k_in  = blockIdx.x * 256 + threadIdx.x;   // 0..1023 (coalesced)
    const int k_out = blockIdx.y;                       // 0..1023
    const float* wi = w_init + (size_t)k_out * SP;
    float acc = 0.0f;
#pragma unroll 8
    for (int s = 0; s < SP; ++s)
        acc = fmaf(wi[s], w_sample[(size_t)s * KDIM + k_in], acc);
    Mw[(size_t)k_out * KDIM + k_in] = (_Float16)acc;
}

// ---------------------------------------------------------------------------
// Kernel 2: rec[r, n] = sum_k patches[r, k] * M[n, k]  (then fold to image)
// patches row r = (b, nh, nw);  k = (kh, kw): 32 contiguous floats per k-step.
// ---------------------------------------------------------------------------
__device__ __forceinline__ v16h frag16(const _Float16* p) {
    v8h lo = *(const v8h*)p;
    v8h hi = *(const v8h*)(p + 16);
    return __builtin_shufflevector(lo, hi,
        0,1,2,3,4,5,6,7,8,9,10,11,12,13,14,15);
}

__global__ __launch_bounds__(256)
void recon_gemm(const float* __restrict__ x,
                const _Float16* __restrict__ Mw,
                float* __restrict__ y) {
    __shared__ __align__(16) _Float16 As [2][BM * BK];
    __shared__ __align__(16) _Float16 Bsh[2][BN * BK];

    const int tid  = threadIdx.x;
    const int lane = tid & 31;
    const int wave = tid >> 5;
    const int wm   = wave & 3;           // 4 row-groups of 32 rows
    const int wn   = wave >> 2;          // 2 col-groups of 64 cols
    const int rowBase = blockIdx.x * BM;
    const int colBase = blockIdx.y * BN;

    // ---- staging addresses (per thread: 16 elements of A and of B per step)
    const int srow  = tid >> 1;          // 0..127
    const int shalf = tid & 1;           // which 16-wide half of the 32 k's
    const int gr  = rowBase + srow;
    const int gb  = gr >> 10, gp = gr & 1023, gnh = gp >> 5, gnw = gp & 31;
    // x index: ((b*1024 + nh*32 + kh)*1024) + nw*32 + kw ; kh == ks
    const float* a_src0 = x + ((size_t)((gb << 10) + (gnh << 5)) << 10)
                            + (gnw << 5) + shalf * 16;
    const _Float16* b_src0 = Mw + (size_t)(colBase + srow) * KDIM + shalf * 16;

    const int sdst = srow * BK + shalf * 16;

    // A tile: f32 -> f16 convert through VGPRs into LDS
    auto stageA = [&](int buf, int ks) {
        const float4* as = (const float4*)(a_src0 + (size_t)ks * 1024);
        float4 f0 = as[0], f1 = as[1], f2 = as[2], f3 = as[3];
        v8h lo = { (_Float16)f0.x, (_Float16)f0.y, (_Float16)f0.z, (_Float16)f0.w,
                   (_Float16)f1.x, (_Float16)f1.y, (_Float16)f1.z, (_Float16)f1.w };
        v8h hi = { (_Float16)f2.x, (_Float16)f2.y, (_Float16)f2.z, (_Float16)f2.w,
                   (_Float16)f3.x, (_Float16)f3.y, (_Float16)f3.z, (_Float16)f3.w };
        _Float16* ad = &As[buf][sdst];
        *(v8h*)ad       = lo;
        *(v8h*)(ad + 8) = hi;
    };

    // B tile: fp16 weights copied global -> LDS with the CDNA5 async data path
    // (no VGPR round trip; tracked on ASYNCcnt). IOFFSET applies to both the
    // LDS and the global address, so offset:16 covers the second 16B chunk.
    auto stageB = [&](int buf, int ks) {
        unsigned lds0 = (unsigned)(size_t)&Bsh[buf][sdst];
        unsigned long long g0 =
            (unsigned long long)(size_t)(b_src0 + (size_t)ks * BK);
        asm volatile("global_load_async_to_lds_b128 %0, %1, off"
                     :: "v"(lds0), "v"(g0) : "memory");
        asm volatile("global_load_async_to_lds_b128 %0, %1, off offset:16"
                     :: "v"(lds0), "v"(g0) : "memory");
    };

    v8f acc[2][4] = {};

    const int fr = lane & 15;            // fragment row (M or N)
    const int fk = (lane >> 4) << 3;     // K sub-chunk: 0 or 8 (+16 for hi half)

    stageB(0, 0);
    stageA(0, 0);
    asm volatile("s_wait_asynccnt 0" ::: "memory");
    __syncthreads();

    for (int ks = 0; ks < KDIM / BK; ++ks) {
        const int buf = ks & 1;
        if (ks + 1 < KDIM / BK) {
            stageB(buf ^ 1, ks + 1);     // async copy overlaps with compute
            stageA(buf ^ 1, ks + 1);
        }

        const _Float16* at = &As [buf][0];
        const _Float16* bt = &Bsh[buf][0];

        const v16h a0 = frag16(at + (wm * 32 + 0  + fr) * BK + fk);
        const v16h a1 = frag16(at + (wm * 32 + 16 + fr) * BK + fk);
#pragma unroll
        for (int tn = 0; tn < 4; ++tn) {
            const v16h bfr = frag16(bt + (wn * 64 + tn * 16 + fr) * BK + fk);
            acc[0][tn] = __builtin_amdgcn_wmma_f32_16x16x32_f16(
                false, a0, false, bfr, (short)0, acc[0][tn], false, false);
            acc[1][tn] = __builtin_amdgcn_wmma_f32_16x16x32_f16(
                false, a1, false, bfr, (short)0, acc[1][tn], false, false);
        }
        // Drain this wave's async copies before the workgroup barrier
        // (the barrier itself does not wait on ASYNCcnt).
        asm volatile("s_wait_asynccnt 0" ::: "memory");
        __syncthreads();
    }

    // ---- epilogue: C layout (lane&15 = N, lane>>4 selects M+8, VGPR v = M row)
    // Output is written once and never re-read: use non-temporal stores so it
    // does not evict the L2-resident x (64MB) / M (2MB) working set.
    const int half = lane >> 4;
    const int cn   = lane & 15;
#pragma unroll
    for (int tm = 0; tm < 2; ++tm) {
#pragma unroll
        for (int tn = 0; tn < 4; ++tn) {
            const int colT = colBase + wn * 64 + tn * 16 + cn;
            const int kh = colT >> 5, kw = colT & 31;
#pragma unroll
            for (int v = 0; v < 8; ++v) {
                const int r  = rowBase + wm * 32 + tm * 16 + half * 8 + v;
                const int b  = r >> 10, p = r & 1023;
                const int nh = p >> 5,  nw = p & 31;
                const size_t idx =
                    ((size_t)((b << 10) + (nh << 5) + kh) << 10) + (nw << 5) + kw;
                __builtin_nontemporal_store(acc[tm][tn][v], &y[idx]);
            }
        }
    }
}

// ---------------------------------------------------------------------------
extern "C" void kernel_launch(void* const* d_in, const int* in_sizes, int n_in,
                              void* d_out, int out_size, void* d_ws, size_t ws_size,
                              hipStream_t stream) {
    (void)in_sizes; (void)n_in; (void)out_size; (void)ws_size;
    const float* x        = (const float*)d_in[0];
    const float* w_sample = (const float*)d_in[1];
    const float* w_init   = (const float*)d_in[2];
    float*       y        = (float*)d_out;
    _Float16*    Mw       = (_Float16*)d_ws;   // 1024*1024 fp16 = 2 MB scratch

    build_weight<<<dim3(KDIM / 256, KDIM), 256, 0, stream>>>(w_sample, w_init, Mw);
    recon_gemm <<<dim3(NROWS / BM, KDIM / BN), 256, 0, stream>>>(x, Mw, y);
}
